// GCN_22213570854982
// MI455X (gfx1250) — matrix operand order repaired and checked by reading, not compiled, not run on hardware
//
#include <hip/hip_runtime.h>

// ---------------------------------------------------------------------------
// GCN on MI455X (gfx1250, wave32).
//  - Dense h@W via V_WMMA_F32_16X16X32_BF16 (bf16 mul, f32 accumulate)
//      * A-fragments preloaded once per wave (A read exactly once per layer)
//      * W staged in LDS (<=64KB of the 320KB WGP LDS), B-frags via ds_load
//  - Edge aggregation via L2-resident float atomics + global_prefetch
//  - BatchNorm via block-partial sums + per-feature finalize
// ---------------------------------------------------------------------------

typedef __bf16 bf16;
typedef __attribute__((ext_vector_type(16))) __bf16 v16bf;
typedef __attribute__((ext_vector_type(8)))  float  v8f;

#define GENE_DIM 256
#define HID      128
#define BN_EPS   1e-5f

// -------------------------------- utilities --------------------------------

__global__ void k_fill(float* __restrict__ p, float v, long long n) {
  long long i = (long long)blockIdx.x * blockDim.x + threadIdx.x;
  long long stride = (long long)gridDim.x * blockDim.x;
  for (; i < n; i += stride) p[i] = v;
}

__global__ void k_deg_edges(const int* __restrict__ dst, float* __restrict__ deg, int E) {
  int i = blockIdx.x * blockDim.x + threadIdx.x;
  int stride = gridDim.x * blockDim.x;
  for (; i < E; i += stride) atomicAdd(&deg[dst[i]], 1.0f);
}

__global__ void k_dinv(float* __restrict__ deg, int n) {
  int i = blockIdx.x * blockDim.x + threadIdx.x;
  if (i < n) deg[i] = rsqrtf(fmaxf(deg[i], 1.0f));
}

// Wt[n][k] = (bf16) W[k][n]   (row-major W is [K][N])
__global__ void k_w_to_bf16_t(const float* __restrict__ W, bf16* __restrict__ Wt,
                              int K, int N) {
  int idx = blockIdx.x * blockDim.x + threadIdx.x;
  if (idx >= K * N) return;
  int nn = idx / K, kk = idx - nn * K;
  Wt[idx] = (bf16)W[kk * N + nn];
}

// ------------------------------- WMMA GEMM ---------------------------------
// C[M,N] = A[M,K](f32 -> bf16 in-reg) @ Wt[N,K](bf16, staged in LDS).
// Block = 8 waves; wave w computes rows [16*(8*blk+w), +16) x ALL N columns:
//   - all K/32 A-fragments held in registers (read A once per layer)
//   - B-fragments read from LDS per N-tile
// M % 16 == 0 (100000 = 6250*16), K in {128,256}, N % 16 == 0.
template <int K>
__global__ void __launch_bounds__(256)
k_gemm_bf16(const float* __restrict__ A, const bf16* __restrict__ Wt,
            float* __restrict__ C, int M, int N) {
  extern __shared__ bf16 sW[];
  constexpr int NK = K / 32;

  {  // stage Wt (N*K bf16) into LDS, coalesced 16B per thread
    const uint4* s = (const uint4*)Wt;
    uint4* d = (uint4*)sW;
    const int nv = (N * K) / 8;
    for (int i = threadIdx.x; i < nv; i += 256) d[i] = s[i];
  }
  __syncthreads();

  const int wave = threadIdx.x >> 5;
  const int lane = threadIdx.x & 31;
  const int half = lane >> 4;   // lanes 0-15 vs 16-31 (wave32 fragment halves)
  const int l15  = lane & 15;
  const int tileM = blockIdx.x * 8 + wave;
  if (tileM * 16 >= M) return;

  // Preload all A fragments for this wave's 16 rows.
  // A k order used here (contiguous, same as B): lanes 0-15 -> K {0..15}, lanes 16-31 -> {16..31}
  const float* arow = A + (size_t)(tileM * 16 + l15) * K;
  v16bf afrag[NK];
#pragma unroll
  for (int kk = 0; kk < NK; ++kk) {
    const int k0 = kk * 32;
    const float4 a0 = *(const float4*)(arow + k0 + half * 16);
    const float4 a1 = *(const float4*)(arow + k0 + half * 16 + 4);
    const float4 a2 = *(const float4*)(arow + k0 + half * 16 + 8);
    const float4 a3 = *(const float4*)(arow + k0 + half * 16 + 12);
    v16bf a;
    a[0]=(bf16)a0.x;  a[1]=(bf16)a0.y;  a[2]=(bf16)a0.z;  a[3]=(bf16)a0.w;
    a[4]=(bf16)a1.x;  a[5]=(bf16)a1.y;  a[6]=(bf16)a1.z;  a[7]=(bf16)a1.w;
    a[8]=(bf16)a2.x;  a[9]=(bf16)a2.y;  a[10]=(bf16)a2.z; a[11]=(bf16)a2.w;
    a[12]=(bf16)a3.x; a[13]=(bf16)a3.y; a[14]=(bf16)a3.z; a[15]=(bf16)a3.w;
    afrag[kk] = a;
  }

  const int ntiles = N >> 4;
  for (int tn = 0; tn < ntiles; ++tn) {
    // B k order used here (same as A): lane holds 16 contiguous K of column l15 (+16 for upper half)
    const bf16* brow = sW + (size_t)(tn * 16 + l15) * K + half * 16;
    v8f c = {};
#pragma unroll
    for (int kk = 0; kk < NK; ++kk) {
      const v16bf b = *(const v16bf*)(brow + kk * 32);
      c = __builtin_amdgcn_wmma_f32_16x16x32_bf16(
          /*neg_a=*/false, afrag[kk], /*neg_b=*/false, b,
          /*c_mod=*/(short)0, c, /*reuse_a=*/false, /*reuse_b=*/false);
    }
    // C/D layout: VGPR r -> row half*8 + r, col l15
    float* crow = C + ((size_t)tileM * 16 + half * 8) * N + tn * 16 + l15;
#pragma unroll
    for (int r = 0; r < 8; ++r) crow[(size_t)r * N] = c[r];
  }
}

// --------------------------- edge scatter-add ------------------------------
// agg[dst] += m[src] * dinv[src]*dinv[dst]; each thread owns one float4 of one edge.
__global__ void k_edge_scatter(const int* __restrict__ src, const int* __restrict__ dst,
                               const float* __restrict__ dinv, const float* __restrict__ m,
                               float* __restrict__ agg, int d, int qbits,
                               long long total) {
  const int qmask = (d >> 2) - 1;
  long long tid = (long long)blockIdx.x * blockDim.x + threadIdx.x;
  long long stride = (long long)gridDim.x * blockDim.x;
  for (; tid < total; tid += stride) {
    // prefetch next iteration's gather row (random-access, latency-bound)
    const long long tnext = tid + stride;
    if (tnext < total) {
      const long long e2 = tnext >> qbits;
      __builtin_prefetch(m + (size_t)src[e2] * d + ((int)tnext & qmask) * 4, 0, 0);
    }
    const long long e = tid >> qbits;
    const int q = (int)tid & qmask;
    const int s = src[e], t = dst[e];
    const float nrm = dinv[s] * dinv[t];
    const float4 mv = ((const float4*)(m + (size_t)s * d))[q];
    float* ap = agg + (size_t)t * d + q * 4;
    atomicAdd(ap + 0, mv.x * nrm);
    atomicAdd(ap + 1, mv.y * nrm);
    atomicAdd(ap + 2, mv.z * nrm);
    atomicAdd(ap + 3, mv.w * nrm);
  }
}

// out = relu(agg + m*dinv^2 (self-loop) + bias)   [float4 streaming]
__global__ void k_self_bias_relu(const float4* __restrict__ agg, const float4* __restrict__ m,
                                 const float* __restrict__ dinv, const float4* __restrict__ b,
                                 float4* __restrict__ out, int qbits, int qmask,
                                 long long totalq) {
  long long idx = (long long)blockIdx.x * blockDim.x + threadIdx.x;
  long long stride = (long long)gridDim.x * blockDim.x;
  for (; idx < totalq; idx += stride) {
    const int node = (int)(idx >> qbits);
    const int q = (int)idx & qmask;
    const float di = dinv[node];
    const float d2 = di * di;
    const float4 av = agg[idx], mv = m[idx], bv = b[q];
    float4 r;
    r.x = fmaxf(av.x + mv.x * d2 + bv.x, 0.0f);
    r.y = fmaxf(av.y + mv.y * d2 + bv.y, 0.0f);
    r.z = fmaxf(av.z + mv.z * d2 + bv.z, 0.0f);
    r.w = fmaxf(av.w + mv.w * d2 + bv.w, 0.0f);
    out[idx] = r;
  }
}

// ------------------------------- BatchNorm ---------------------------------

__global__ void k_bn_stats(const float* __restrict__ h, float* __restrict__ sum,
                           float* __restrict__ sumsq, int n, int d) {
  const int f = threadIdx.x;  // blockDim.x == d
  float s = 0.f, ss = 0.f;
  for (int i = blockIdx.x; i < n; i += gridDim.x) {
    const float v = h[(size_t)i * d + f];
    s += v; ss += v * v;
  }
  atomicAdd(&sum[f], s);
  atomicAdd(&sumsq[f], ss);
}

__global__ void k_bn_scale(const float* __restrict__ sum, const float* __restrict__ sumsq,
                           const float* __restrict__ gamma, const float* __restrict__ beta,
                           float* __restrict__ scale, float* __restrict__ shift,
                           int d, float inv_n) {
  int f = threadIdx.x;
  if (f >= d) return;
  const float mean = sum[f] * inv_n;
  const float var = sumsq[f] * inv_n - mean * mean;
  const float s = gamma[f] * rsqrtf(var + BN_EPS);
  scale[f] = s;
  shift[f] = beta[f] - mean * s;
}

__global__ void k_bn_apply(const float4* __restrict__ h, const float4* __restrict__ scale,
                           const float4* __restrict__ shift, float4* __restrict__ out,
                           int qmask, long long totalq) {
  long long idx = (long long)blockIdx.x * blockDim.x + threadIdx.x;
  long long stride = (long long)gridDim.x * blockDim.x;
  for (; idx < totalq; idx += stride) {
    const int q = (int)idx & qmask;
    const float4 hv = h[idx], sc = scale[q], sh = shift[q];
    float4 r;
    r.x = hv.x * sc.x + sh.x;
    r.y = hv.y * sc.y + sh.y;
    r.z = hv.z * sc.z + sh.z;
    r.w = hv.w * sc.w + sh.w;
    out[idx] = r;
  }
}

// --------------------------- classifier (64 -> 2) --------------------------
__global__ void k_classifier(const float* __restrict__ bn4, const float* __restrict__ Wc,
                             const float* __restrict__ bc, float* __restrict__ logits,
                             int n) {
  int i = blockIdx.x * blockDim.x + threadIdx.x;
  if (i >= n) return;
  float a0 = bc[0], a1 = bc[1];
  const float* r = bn4 + (size_t)i * 64;
#pragma unroll
  for (int k = 0; k < 64; ++k) {
    const float v = r[k];
    a0 += v * Wc[2 * k];
    a1 += v * Wc[2 * k + 1];
  }
  logits[2 * i + 0] = a0;
  logits[2 * i + 1] = a1;
}

// ------------------------------- launcher ----------------------------------

extern "C" void kernel_launch(void* const* d_in, const int* in_sizes, int n_in,
                              void* d_out, int out_size, void* d_ws, size_t ws_size,
                              hipStream_t stream) {
  (void)n_in; (void)out_size; (void)ws_size;

  const float* x   = (const float*)d_in[0];
  const int*   ei  = (const int*)d_in[1];
  const int    E   = in_sizes[1] / 2;
  const int*   src = ei;       // edge_index[0]
  const int*   dst = ei + E;   // edge_index[1]
  const int    n   = in_sizes[0] / GENE_DIM;

  const float* Wl[4]  = {(const float*)d_in[2],  (const float*)d_in[4],
                         (const float*)d_in[6],  (const float*)d_in[8]};
  const float* bl[4]  = {(const float*)d_in[3],  (const float*)d_in[5],
                         (const float*)d_in[7],  (const float*)d_in[9]};
  const float* Wc = (const float*)d_in[10];
  const float* bc = (const float*)d_in[11];
  const float* gl[4]  = {(const float*)d_in[12], (const float*)d_in[14],
                         (const float*)d_in[16], (const float*)d_in[18]};
  const float* bel[4] = {(const float*)d_in[13], (const float*)d_in[15],
                         (const float*)d_in[17], (const float*)d_in[19]};

  const int Kd[4] = {GENE_DIM, HID, 2 * HID, HID};
  const int Nd[4] = {HID, 2 * HID, HID, HID / 2};

  // workspace layout (~205.4 MB): two ping-pong [n,256] f32 buffers + small scratch
  char* w = (char*)d_ws;
  float* bufA = (float*)w; w += (size_t)n * 256 * sizeof(float);
  float* bufB = (float*)w; w += (size_t)n * 256 * sizeof(float);
  float* dinv = (float*)w; w += (((size_t)n * sizeof(float)) + 255) & ~(size_t)255;
  bf16*  wt   = (bf16*)w;  w += 65536 * sizeof(bf16);
  float* sum   = (float*)w; w += 256 * sizeof(float);
  float* sumsq = (float*)w; w += 256 * sizeof(float);
  float* scale = (float*)w; w += 256 * sizeof(float);
  float* shift = (float*)w; w += 256 * sizeof(float);

  float* logits = (float*)d_out;
  float* out4   = logits + (size_t)2 * n;
  float* bn4    = out4 + (size_t)64 * n;

  // ---- degrees -> dinv (self-loop contributes 1 to every node's degree) ----
  k_fill<<<256, 256, 0, stream>>>(dinv, 1.0f, n);
  k_deg_edges<<<2048, 256, 0, stream>>>(dst, dinv, E);
  k_dinv<<<(n + 255) / 256, 256, 0, stream>>>(dinv, n);

  const float* hin = x;
  for (int l = 0; l < 4; ++l) {
    const int K = Kd[l], N = Nd[l];
    const int dbits = __builtin_ctz(N);
    const long long nd = (long long)n * N;
    const long long ndq = nd >> 2;
    const int qmask = (N >> 2) - 1;

    float *m, *agg;
    if (l == 0) { m = bufA; agg = bufB; }
    else        { m = (hin == bufA) ? bufB : bufA; agg = (float*)hin; }
    float* post  = (l == 3) ? out4 : agg;  // relu output (in-place over agg unless last)
    float* hnext = (l == 3) ? bn4  : m;    // BN output (in-place over m unless last)

    // 1) W -> bf16 transposed [N][K]
    k_w_to_bf16_t<<<(K * N + 255) / 256, 256, 0, stream>>>(Wl[l], wt, K, N);

    // 2) m = hin @ W  (WMMA bf16 -> f32; W in LDS, A read once)
    const dim3 ggrid((n / 16 + 7) / 8);
    const size_t lds = (size_t)K * N * sizeof(bf16);
    if (K == 256)
      k_gemm_bf16<256><<<ggrid, 256, lds, stream>>>(hin, wt, m, n, N);
    else
      k_gemm_bf16<128><<<ggrid, 256, lds, stream>>>(hin, wt, m, n, N);

    // 3) agg = 0; scatter-add normalized messages (L2-resident atomics)
    k_fill<<<4096, 256, 0, stream>>>(agg, 0.0f, nd);
    k_edge_scatter<<<8192, 256, 0, stream>>>(src, dst, dinv, m, agg, N, dbits - 2,
                                             (long long)E << (dbits - 2));

    // 4) add self-loop + bias, relu
    k_self_bias_relu<<<8192, 256, 0, stream>>>((const float4*)agg, (const float4*)m,
                                               dinv, (const float4*)bl[l], (float4*)post,
                                               dbits - 2, qmask, ndq);

    // 5) BatchNorm (training mode, biased variance)
    k_fill<<<2, 256, 0, stream>>>(sum, 0.0f, 512);  // zero sum+sumsq (contiguous)
    k_bn_stats<<<512, N, 0, stream>>>(post, sum, sumsq, n, N);
    k_bn_scale<<<1, 256, 0, stream>>>(sum, sumsq, gl[l], bel[l], scale, shift,
                                      N, 1.0f / (float)n);
    k_bn_apply<<<8192, 256, 0, stream>>>((const float4*)post, (const float4*)scale,
                                         (const float4*)shift, (float4*)hnext,
                                         qmask, ndq);

    hin = hnext;
  }

  // logits = bn4 @ Wc + bc
  k_classifier<<<(n + 255) / 256, 256, 0, stream>>>(bn4, Wc, bc, logits, n);
}
